// MultiPillarCounter_7954279432889
// MI455X (gfx1250) — compile-verified, matrix operand order checked by reading, not run on hardware
//
#include <hip/hip_runtime.h>
#include <stdint.h>

typedef __attribute__((ext_vector_type(16))) _Float16 v16h;
typedef __attribute__((ext_vector_type(8)))  float    v8f;

// ---------------- workspace bitmap layout (u32 words) ----------------
// res0: 2048x2048 bits, res1: 1024x1024, res2: 512x512
static constexpr unsigned W0   = 2048u * 2048u / 32u;   // 131072 words
static constexpr unsigned W1   = 1024u * 1024u / 32u;   //  32768 words
static constexpr unsigned W2   =  512u *  512u / 32u;   //   8192 words
static constexpr unsigned OFF0 = 0u;
static constexpr unsigned OFF1 = OFF0 + W0;
static constexpr unsigned OFF2 = OFF1 + W1;
static constexpr unsigned WTOT = OFF2 + W2;             // 172032 words = 688128 B

// ---------------- kernel 1: zero the bitmaps ----------------
__global__ void mpc_zero_ws(unsigned int* __restrict__ ws) {
    unsigned i = blockIdx.x * blockDim.x + threadIdx.x;
    if (i < WTOT) ws[i] = 0u;
}

// ---------------- kernel 2: scatter points into bitmaps ----------------
__device__ __forceinline__ void mpc_setbit(unsigned int* __restrict__ bits,
                                           float px, float py,
                                           float psx, float psy,
                                           int G, unsigned off) {
    // match reference: floor(pt / pillar_size), clip to [0, G-1]
    int cx = (int)floorf(px / psx);
    int cy = (int)floorf(py / psy);
    cx = min(max(cx, 0), G - 1);
    cy = min(max(cy, 0), G - 1);
    unsigned idx = (unsigned)cy * (unsigned)G + (unsigned)cx;   // grid[cy, cx]
    atomicOr(&bits[off + (idx >> 5)], 1u << (idx & 31u));       // no-return atomic
}

__global__ void mpc_scatter(const float* __restrict__ pts,
                            const float* __restrict__ ps,     // [3][2]
                            const float* __restrict__ pcmin,  // [2]
                            unsigned int* __restrict__ bits,
                            int npoints) {
    const float minx = pcmin[0], miny = pcmin[1];
    const float p0x = ps[0], p0y = ps[1];
    const float p1x = ps[2], p1y = ps[3];
    const float p2x = ps[4], p2y = ps[5];
    const int stride = gridDim.x * blockDim.x;
    const float2* __restrict__ pxy = (const float2*)pts;
    for (int i = blockIdx.x * blockDim.x + threadIdx.x; i < npoints; i += stride) {
        float2 p = pxy[i];                       // global_load_b64, coalesced
        float px = p.x - minx;
        float py = p.y - miny;
        mpc_setbit(bits, px, py, p0x, p0y, 2048, OFF0);
        mpc_setbit(bits, px, py, p1x, p1y, 1024, OFF1);
        mpc_setbit(bits, px, py, p2x, p2y,  512, OFF2);
    }
}

// ---------------- kernel 3: per-slice popcount reduction ----------------
// 112 blocks x 256 threads; block b -> one output slice.
__global__ void mpc_reduce(const unsigned int* __restrict__ bits,
                           float* __restrict__ out) {
    __shared__ unsigned int lbits[2048];   // max slice = 2048 words (8 KB)
    __shared__ float wsum[8];

    const int b = blockIdx.x;
    const int t = threadIdx.x;
    unsigned base, wordsPerSlice;
    if (b < 64)      { base = OFF0 + (unsigned)(b)      * 2048u; wordsPerSlice = 2048u; }
    else if (b < 96) { base = OFF1 + (unsigned)(b - 64) * 1024u; wordsPerSlice = 1024u; }
    else             { base = OFF2 + (unsigned)(b - 96) *  512u; wordsPerSlice =  512u; }
    const unsigned wpt = wordsPerSlice >> 8;   // words per thread: 8 / 4 / 2

    int p = 0;
#if __has_builtin(__builtin_amdgcn_global_load_async_to_lds_b32)
    // CDNA5 async global->LDS path (ASYNCcnt)
    for (unsigned k = 0; k < wpt; ++k) {
        unsigned w = (unsigned)t + (k << 8);
        __builtin_amdgcn_global_load_async_to_lds_b32(
            (__attribute__((address_space(1))) int*)(bits + base + w),
            (__attribute__((address_space(3))) int*)&lbits[w],
            0, 0);
    }
    asm volatile("s_wait_asynccnt 0" ::: "memory");
    __syncthreads();
    for (unsigned k = 0; k < wpt; ++k)
        p += __popc(lbits[(unsigned)t + (k << 8)]);
#else
    for (unsigned k = 0; k < wpt; ++k)
        p += __popc(bits[base + (unsigned)t + (k << 8)]);
    (void)lbits;
#endif

    // ---- wave-level exact sum of 32 per-lane partials via WMMA ----
    // Each lane splats p (<=256, exact in f16) into all 16 A slots; B = ones.
    // Row M of A sums to 16*(p_M + p_{M+16}); with the C/D layout,
    // sum_j c[j] = 16*Sum_{M=0..7}(p_M+p_{M+16}) on lanes 0-15 and the
    // M=8..15 half on lanes 16-31. (s0 + s16)/16 == exact 32-lane sum.
    _Float16 hp = (_Float16)(float)p;
    v16h a, ones;
#pragma unroll
    for (int i = 0; i < 16; ++i) { a[i] = hp; ones[i] = (_Float16)1.0f; }
    v8f c = {};
    c = __builtin_amdgcn_wmma_f32_16x16x32_f16(
            /*neg_a=*/false, a, /*neg_b=*/false, ones,
            /*c_mod=*/(short)0, c, /*reuse_a=*/false, /*reuse_b=*/false);
    float s = c[0] + c[1] + c[2] + c[3] + c[4] + c[5] + c[6] + c[7];
    float waveSum = (__shfl(s, 0) + __shfl(s, 16)) * 0.0625f;

    if ((t & 31) == 0) wsum[t >> 5] = waveSum;
    __syncthreads();
    if (t == 0) {
        float tot = 0.0f;
#pragma unroll
        for (int i = 0; i < 8; ++i) tot += wsum[i];
        out[b] = tot;
    }
}

// ---------------- launcher ----------------
extern "C" void kernel_launch(void* const* d_in, const int* in_sizes, int n_in,
                              void* d_out, int out_size, void* d_ws, size_t ws_size,
                              hipStream_t stream) {
    const float* pts   = (const float*)d_in[0];   // [N,2] f32
    const float* ps    = (const float*)d_in[1];   // [3,2] f32
    const float* pcmin = (const float*)d_in[2];   // [2]   f32
    unsigned int* bits = (unsigned int*)d_ws;     // 688128 bytes used
    float* out = (float*)d_out;                   // 112 f32
    const int npoints = in_sizes[0] / 2;

    mpc_zero_ws<<<(WTOT + 255u) / 256u, 256, 0, stream>>>(bits);
    mpc_scatter<<<4096, 256, 0, stream>>>(pts, ps, pcmin, bits, npoints);
    mpc_reduce<<<112, 256, 0, stream>>>(bits, out);
}
